// CAWapper_987842478479
// MI455X (gfx1250) — matrix operand order repaired and checked by reading, not compiled, not run on hardware
//
#include <hip/hip_runtime.h>
#include <stdint.h>

// ---------------- problem constants ----------------
#define NB   8
#define CF   256
#define HF   64
#define WF   64
#define HD   32
#define LP   1024          // HD*WD patches / spatial positions
#define KD   2304          // CF*3*3 patch length (divisible by 32)

// ---------------- WMMA helper types ----------------
typedef __attribute__((ext_vector_type(16))) __bf16       v16bf;
typedef __attribute__((ext_vector_type(8)))  float        v8f;
typedef __attribute__((ext_vector_type(4)))  unsigned int v4u;

union Frag16 { v16bf v; v4u q[2]; unsigned short s[16]; };
union AccF   { v8f   v; float f[8]; };

__device__ __forceinline__ unsigned short f2bf(float f) {
  unsigned int u = __float_as_uint(f);
  return (unsigned short)((u + 0x7FFFu + ((u >> 16) & 1u)) >> 16); // RNE
}

__device__ __forceinline__ v8f wmma_bf16(Frag16 a, Frag16 b, v8f c) {
  // (neg_a, A, neg_b, B, c_mod, C, reuse_a, reuse_b)
  return __builtin_amdgcn_wmma_f32_16x16x32_bf16(false, a.v, false, b.v,
                                                 (short)0, c, false, false);
}

// async 16B global -> LDS copy (ASYNCcnt-tracked, bypasses VGPRs).
// ldsOff = low 32 bits of generic shared pointer (LDS aperture truncation).
__device__ __forceinline__ void async_g2l_b128(unsigned ldsOff, const void* gaddr) {
  asm volatile("global_load_async_to_lds_b128 %0, %1, off"
               :: "v"(ldsOff), "v"(gaddr) : "memory");
}
__device__ __forceinline__ void wait_async0() {
  asm volatile("s_wait_asynccnt 0x0" ::: "memory");
}

__device__ __forceinline__ int trflat(int i) { return ((i & 31) << 5) | (i >> 5); }

// ---------------- 1) feature -> bf16 ----------------
__global__ void k_f2bf(const float* __restrict__ f, unsigned short* __restrict__ fbf) {
  int i = blockIdx.x * 256 + threadIdx.x;           // NB*CF*HF*WF threads
  fbf[i] = f2bf(f[i]);
}

// ---------------- 2) mask -> per-patch validity mm ----------------
__global__ void k_mask(const float* __restrict__ mask, float* __restrict__ mm) {
  int q = blockIdx.x * 256 + threadIdx.x;           // 1024
  if (q >= LP) return;
  int qh = q >> 5, qw = q & 31;
  int bad = 0;
  for (int di = -1; di <= 1; ++di)
    for (int dj = -1; dj <= 1; ++dj) {
      int h = qh + di, w = qw + dj;
      if (h < 0 || h >= HD || w < 0 || w >= HD) continue;
      const float* b = mask + (4 * h) * 128 + 4 * w;   // batch 0 only (reference)
      float s4 = b[0] + b[1] + b[128] + b[129];
      if (s4 > 2.0f) bad = 1;                           // round-half-even of s4/4 != 0
    }
  mm[q] = bad ? 0.0f : 1.0f;
}

// ---------------- 3) bf16 patches of downsampled feature + 1/norm ----------------
__global__ void k_patches(const float* __restrict__ f, unsigned short* __restrict__ Wbf,
                          float* __restrict__ invn) {
  __shared__ float red[256];
  int p = blockIdx.x, n = blockIdx.y, tid = threadIdx.x;
  int qh = p >> 5, qw = p & 31;
  float s = 0.0f;
  for (int k = tid; k < KD; k += 256) {               // 9 iterations
    int c = k / 9, r = k % 9, di = r / 3, dj = r % 3;
    int h0 = qh + di - 1, w0 = qw + dj - 1;
    float v = 0.0f;
    if (h0 >= 0 && h0 < HD && w0 >= 0 && w0 < HD)
      v = f[((size_t)(n * CF + c) * HF + 2 * h0) * WF + 2 * w0];
    Wbf[(size_t)(n * LP + p) * KD + k] = f2bf(v);
    s += v * v;
  }
  red[tid] = s; __syncthreads();
  for (int st = 128; st > 0; st >>= 1) { if (tid < st) red[tid] += red[tid + st]; __syncthreads(); }
  if (tid == 0) invn[n * LP + p] = 1.0f / fmaxf(sqrtf(red[0]), 1e-4f);
}

// ---------------- 4) similarity GEMM: yi[q,p] = invn[q] * <W[q], W[p]> ----------------
// block tile 128x128, K-step 32, 8 waves (4 M-groups x 2 N-groups), wave = 32M x 64N
#define LDSS 40
__global__ void k_gemm_sim(const unsigned short* __restrict__ Wbf,
                           const float* __restrict__ invn, float* __restrict__ yi) {
  __shared__ unsigned short As[128 * LDSS];
  __shared__ unsigned short Bs[128 * LDSS];
  int tid = threadIdx.x, lane = tid & 31, wid = tid >> 5;
  int hi = lane >> 4, lm = lane & 15;
  int n = blockIdx.z;
  int qBase = blockIdx.y * 128, pBase = blockIdx.x * 128;
  int waveM = (wid >> 1) * 32, waveN = (wid & 1) * 64;

  AccF acc[2][4];
  for (int mi = 0; mi < 2; ++mi)
    for (int ni = 0; ni < 4; ++ni)
      for (int r = 0; r < 8; ++r) acc[mi][ni].f[r] = 0.0f;

  for (int kb = 0; kb < KD; kb += 32) {
    for (int it = 0; it < 2; ++it) {
      int chunk = tid + it * 256, row = chunk >> 2, sub = chunk & 3;
      const unsigned short* gA = Wbf + (size_t)(n * LP + qBase + row) * KD + kb + sub * 8;
      const unsigned short* gB = Wbf + (size_t)(n * LP + pBase + row) * KD + kb + sub * 8;
      async_g2l_b128((unsigned)(uintptr_t)&As[row * LDSS + sub * 8], gA);
      async_g2l_b128((unsigned)(uintptr_t)&Bs[row * LDSS + sub * 8], gB);
      if (kb + 32 < KD) {                             // prefetch next K-tile (WGP scope)
        __builtin_prefetch((const char*)gA + 64, 0, 3);
        __builtin_prefetch((const char*)gB + 64, 0, 3);
      }
    }
    wait_async0();
    __syncthreads();

    Frag16 af[2], bfr[4];
    for (int mi = 0; mi < 2; ++mi) {
      int base = (waveM + mi * 16 + lm) * LDSS;
      af[mi].q[0] = *(v4u*)&As[base + 8 * hi];        // k = i + 8*hi           (i<8)
      af[mi].q[1] = *(v4u*)&As[base + 16 + 8 * hi];   // k = i + 8 + 8*hi       (i>=8)
    }
    for (int ni = 0; ni < 4; ++ni) {
      int base = (waveN + ni * 16 + lm) * LDSS;
      bfr[ni].q[0] = *(v4u*)&Bs[base + 8 * hi];       // k = i + 8*hi           (i<8)
      bfr[ni].q[1] = *(v4u*)&Bs[base + 16 + 8 * hi];
    }
    for (int mi = 0; mi < 2; ++mi)
      for (int ni = 0; ni < 4; ++ni)
        acc[mi][ni].v = wmma_bf16(af[mi], bfr[ni], acc[mi][ni].v);
    __syncthreads();
  }

  for (int mi = 0; mi < 2; ++mi)
    for (int r = 0; r < 8; ++r) {
      int q = qBase + waveM + mi * 16 + r + 8 * hi;
      float sc = invn[n * LP + q];
      for (int ni = 0; ni < 4; ++ni) {
        int p = pBase + waveN + ni * 16 + lm;
        yi[(size_t)(n * LP + q) * LP + p] = acc[mi][ni].f[r] * sc;
      }
    }
}

// ---------------- 5) fuse pass 1: diagonal 3-tap on flat (q,p) ----------------
__global__ void k_fuse1(const float* __restrict__ in, float* __restrict__ out) {
  int j = blockIdx.x * 256 + threadIdx.x;
  int i = blockIdx.y, n = blockIdx.z;
  const float* b = in + (size_t)n * LP * LP;
  float v = b[(size_t)i * LP + j];
  if (i > 0 && j > 0)           v += b[(size_t)(i - 1) * LP + j - 1];
  if (i < LP - 1 && j < LP - 1) v += b[(size_t)(i + 1) * LP + j + 1];
  out[(size_t)n * LP * LP + (size_t)i * LP + j] = v;
}

// ---------------- 6) fuse pass 2: same op in (w,h)-transposed flat indexing ----------------
__global__ void k_fuse2(const float* __restrict__ in, float* __restrict__ out) {
  int p = blockIdx.x * 256 + threadIdx.x;
  int q = blockIdx.y, n = blockIdx.z;
  int q2 = trflat(q), p2 = trflat(p);
  const float* b = in + (size_t)n * LP * LP;
  float v = 0.0f;
  for (int d = -1; d <= 1; ++d) {
    int a = q2 + d, c = p2 + d;
    if (a >= 0 && a < LP && c >= 0 && c < LP)
      v += b[(size_t)trflat(a) * LP + trflat(c)];
  }
  out[(size_t)n * LP * LP + (size_t)q * LP + p] = v;
}

// ---------------- 7) mask*softmax(axis=q)*mask -> bf16, stored transposed [p][q] ----------------
__global__ void k_softmax(const float* __restrict__ yi, const float* __restrict__ mm,
                          unsigned short* __restrict__ attnbf) {
  __shared__ float smm[LP];
  __shared__ float red[4][64];
  int tid = threadIdx.x, tx = tid & 63, ty = tid >> 6;
  int n = blockIdx.y, p = blockIdx.x * 64 + tx;
  for (int i = tid; i < LP; i += 256) smm[i] = mm[i];
  __syncthreads();

  float mx = -1e30f;
  for (int q = ty; q < LP; q += 4)
    mx = fmaxf(mx, yi[(size_t)(n * LP + q) * LP + p] * smm[q] * 10.0f);
  red[ty][tx] = mx; __syncthreads();
  mx = fmaxf(fmaxf(red[0][tx], red[1][tx]), fmaxf(red[2][tx], red[3][tx]));
  __syncthreads();

  float s = 0.0f;
  for (int q = ty; q < LP; q += 4)
    s += expf(yi[(size_t)(n * LP + q) * LP + p] * smm[q] * 10.0f - mx);
  red[ty][tx] = s; __syncthreads();
  s = red[0][tx] + red[1][tx] + red[2][tx] + red[3][tx];
  float inv = 1.0f / s;
  __syncthreads();

  for (int q = ty; q < LP; q += 4) {
    float a = expf(yi[(size_t)(n * LP + q) * LP + p] * smm[q] * 10.0f - mx) * inv * smm[q];
    attnbf[(size_t)(n * LP + p) * LP + q] = f2bf(a);  // transposed: [p][q]
  }
}

// ---------------- 8) deconv GEMM: out[c,y,x] = 0.25 * sum over 4 (tap,shift) combos ----------------
// grid: x = cBlock(4 x 64ch), y = mBlock(16 x 64pos), z = n*4 + parity(ey,ex)
__global__ void k_deconv(const unsigned short* __restrict__ attnbf,
                         const unsigned short* __restrict__ fbf,
                         float* __restrict__ out) {
  __shared__ unsigned short sA[4][64 * LDSS];
  __shared__ unsigned short sB[4][64 * LDSS];
  int tid = threadIdx.x, lane = tid & 31, wid = tid >> 5;
  int hi = lane >> 4, lm = lane & 15;
  int cls = blockIdx.z & 3, n = blockIdx.z >> 2;
  int ey = cls >> 1, ex = cls & 1;
  int mBase = blockIdx.y * 64, cBase = blockIdx.x * 64;

  int daA[2], tIA[2], dbB[2], tJB[2];
  if (ey == 0) { daA[0] = -1; tIA[0] = 3; daA[1] = 0; tIA[1] = 1; }
  else         { daA[0] =  0; tIA[0] = 2; daA[1] = 1; tIA[1] = 0; }
  if (ex == 0) { dbB[0] = -1; tJB[0] = 3; dbB[1] = 0; tJB[1] = 1; }
  else         { dbB[0] =  0; tJB[0] = 2; dbB[1] = 1; tJB[1] = 0; }

  int mSub = wid >> 1, cSub0 = (wid & 1) * 2;         // wave owns 16pos x 32ch
  AccF acc[2];
  for (int t = 0; t < 2; ++t) for (int r = 0; r < 8; ++r) acc[t].f[r] = 0.0f;

  for (int kb = 0; kb < LP; kb += 32) {
    // --- A tiles: shifted attention rows, [pos 64][k 32], async global->LDS ---
    int row = tid >> 2, sub = tid & 3;
    int m = mBase + row, my = m >> 5, mxp = m & 31;
    for (int a = 0; a < 2; ++a)
      for (int bcomb = 0; bcomb < 2; ++bcomb) {
        int c4 = a * 2 + bcomb;
        int py = my + daA[a], px = mxp + dbB[bcomb];
        unsigned ldsOff = (unsigned)(uintptr_t)&sA[c4][row * LDSS + sub * 8];
        if (py >= 0 && py < HD && px >= 0 && px < HD) {
          async_g2l_b128(ldsOff,
              attnbf + (size_t)(n * LP + py * HD + px) * LP + kb + sub * 8);
        } else {
          *(v4u*)&sA[c4][row * LDSS + sub * 8] = (v4u){0u, 0u, 0u, 0u};
        }
      }
    // --- B tiles: tap-gathered raw patches of f, [ch 64][k 32] ---
    {
      int id0 = tid * 8, cL = id0 >> 5, kL0 = id0 & 31;
      int cg = cBase + cL;
      int qb = kb + kL0, qh = qb >> 5, qw0 = qb & 31;
      for (int a = 0; a < 2; ++a)
        for (int bcomb = 0; bcomb < 2; ++bcomb) {
          int c4 = a * 2 + bcomb;
          int fy = 2 * qh + tIA[a] - 1;
          union { v4u v; unsigned short s[8]; } pk;
          if (fy >= 0 && fy < HF) {
            const unsigned short* frow = fbf + ((size_t)(n * CF + cg) * HF + fy) * WF;
            for (int e = 0; e < 8; ++e) {
              int fx = 2 * (qw0 + e) + tJB[bcomb] - 1;
              pk.s[e] = (fx >= 0 && fx < WF) ? frow[fx] : (unsigned short)0;
            }
          } else {
            pk.v = (v4u){0u, 0u, 0u, 0u};
          }
          *(v4u*)&sB[c4][cL * LDSS + kL0] = pk.v;
        }
    }
    wait_async0();
    __syncthreads();

    for (int c4 = 0; c4 < 4; ++c4) {
      Frag16 af, b0, b1;
      int abase = (mSub * 16 + lm) * LDSS;
      af.q[0] = *(v4u*)&sA[c4][abase + 8 * hi];
      af.q[1] = *(v4u*)&sA[c4][abase + 16 + 8 * hi];
      int bb0 = (cSub0 * 16 + lm) * LDSS;
      int bb1 = ((cSub0 + 1) * 16 + lm) * LDSS;
      b0.q[0] = *(v4u*)&sB[c4][bb0 + 8 * hi]; b0.q[1] = *(v4u*)&sB[c4][bb0 + 16 + 8 * hi];
      b1.q[0] = *(v4u*)&sB[c4][bb1 + 8 * hi]; b1.q[1] = *(v4u*)&sB[c4][bb1 + 16 + 8 * hi];
      acc[0].v = wmma_bf16(af, b0, acc[0].v);
      acc[1].v = wmma_bf16(af, b1, acc[1].v);
    }
    __syncthreads();
  }

  for (int t = 0; t < 2; ++t) {
    int cch = cBase + (cSub0 + t) * 16 + lm;
    for (int r = 0; r < 8; ++r) {
      int m = mBase + mSub * 16 + r + 8 * hi;
      int y = 2 * (m >> 5) + ey, x = 2 * (m & 31) + ex;
      out[((size_t)(n * CF + cch) * HF + y) * WF + x] = 0.25f * acc[t].f[r];
    }
  }
}

// ---------------- host launch ----------------
extern "C" void kernel_launch(void* const* d_in, const int* in_sizes, int n_in,
                              void* d_out, int out_size, void* d_ws, size_t ws_size,
                              hipStream_t stream) {
  (void)in_sizes; (void)n_in; (void)out_size; (void)ws_size;
  const float* f    = (const float*)d_in[0];   // (8,256,64,64) f32
  const float* mask = (const float*)d_in[1];   // (8,1,128,128) f32
  float* out = (float*)d_out;                  // (8,256,64,64) f32
  char* ws = (char*)d_ws;

  const size_t OFF_WBF  = 0;                       // 8*1024*2304*2 = 37,748,736
  const size_t OFF_YIA  = 37748736;                // 8*1024*1024*4 = 33,554,432
  const size_t OFF_YIB  = 71303168;                // 33,554,432
  const size_t OFF_ATTN = 104857600;               // 8*1024*1024*2 = 16,777,216
  const size_t OFF_FBF  = 121634816;               // 8*256*64*64*2 = 16,777,216
  const size_t OFF_INVN = 138412032;               // 32 KiB
  const size_t OFF_MM   = 138444800;               // 4 KiB  (total ~138.5 MB)

  unsigned short* Wbf    = (unsigned short*)(ws + OFF_WBF);
  float*          yiA    = (float*)(ws + OFF_YIA);
  float*          yiB    = (float*)(ws + OFF_YIB);
  unsigned short* attnbf = (unsigned short*)(ws + OFF_ATTN);
  unsigned short* fbf    = (unsigned short*)(ws + OFF_FBF);
  float*          invn   = (float*)(ws + OFF_INVN);
  float*          mm     = (float*)(ws + OFF_MM);

  k_f2bf   <<<dim3(NB * CF * HF * WF / 256), 256, 0, stream>>>(f, fbf);
  k_mask   <<<dim3(4), 256, 0, stream>>>(mask, mm);
  k_patches<<<dim3(LP, NB), 256, 0, stream>>>(f, Wbf, invn);
  k_gemm_sim<<<dim3(LP / 128, LP / 128, NB), 256, 0, stream>>>(Wbf, invn, yiA);
  k_fuse1  <<<dim3(LP / 256, LP, NB), 256, 0, stream>>>(yiA, yiB);
  k_fuse2  <<<dim3(LP / 256, LP, NB), 256, 0, stream>>>(yiB, yiA);
  k_softmax<<<dim3(LP / 64, NB), 256, 0, stream>>>(yiA, mm, attnbf);
  k_deconv <<<dim3(CF / 64, LP / 64, NB * 4), 256, 0, stream>>>(attnbf, fbf, out);
}
